// Decoder_73272142069768
// MI455X (gfx1250) — compile-verified, matrix operand order checked by reading, not run on hardware
//
#include <hip/hip_runtime.h>
#include <hip/hip_bf16.h>
#include <math.h>

#define B_ 16
#define T_ 250
#define E_ 512
#define H_ 1024
#define V_ 32000
#define H3 (3 * H_)

typedef __attribute__((ext_vector_type(16))) __bf16 v16bf;
typedef __attribute__((ext_vector_type(8)))  float  v8f;

// ---------- helpers ----------

__device__ __forceinline__ unsigned short f2bf(float f) {
    unsigned int u = __float_as_uint(f);
    u += 0x7fffu + ((u >> 16) & 1u);          // round-to-nearest-even
    return (unsigned short)(u >> 16);
}

// Load a 16x32 bf16 WMMA A/B fragment from row-major [rows, ld] bf16 data.
// CDNA5 16-bit operand packing: lanes 0-15 hold K 0..7 (v0-3) and 16..23 (v4-7),
// lanes 16-31 hold K 8..15 and 24..31. Two contiguous b128 loads per lane.
__device__ __forceinline__ v16bf frag_bf(const unsigned short* __restrict__ p,
                                         int ld, int row0, int k0) {
    const int lane = threadIdx.x & 31;
    const unsigned short* q =
        p + (size_t)(row0 + (lane & 15)) * ld + k0 + ((lane >> 4) << 3);
    union { v16bf v; uint4 u[2]; } t;
    t.u[0] = *(const uint4*)q;
    t.u[1] = *(const uint4*)(q + 16);
    return t.v;
}

__device__ __forceinline__ v8f wmma_bf16(const v16bf& a, const v16bf& b, const v8f& c) {
    return __builtin_amdgcn_wmma_f32_16x16x32_bf16(false, a, false, b,
                                                   (short)0, c, false, false);
}

// ---------- prep kernels ----------

// dst[n*K + k] = bf16(src[k*N + n])   (store weights transposed, bf16)
__global__ void k_cvt_transpose(const float* __restrict__ src,
                                unsigned short* __restrict__ dst,
                                int K, int N) {
    const size_t total = (size_t)K * N;
    for (size_t i = (size_t)blockIdx.x * blockDim.x + threadIdx.x; i < total;
         i += (size_t)gridDim.x * blockDim.x) {
        const int n = (int)(i / K);
        const int k = (int)(i % K);
        dst[i] = f2bf(src[(size_t)k * N + n]);
    }
}

// Flat f32 -> bf16 copy (initial recurrent state)
__global__ void k_cvt_bf16(const float* __restrict__ src,
                           unsigned short* __restrict__ dst, int n) {
    const int i = blockIdx.x * blockDim.x + threadIdx.x;
    if (i < n) dst[i] = f2bf(src[i]);
}

// Embedding gather -> bf16 activations [B*T, E]
__global__ void k_embed(const int* __restrict__ tok,
                        const float* __restrict__ emb,
                        unsigned short* __restrict__ dst) {
    const size_t total = (size_t)B_ * T_ * E_;
    for (size_t i = (size_t)blockIdx.x * blockDim.x + threadIdx.x; i < total;
         i += (size_t)gridDim.x * blockDim.x) {
        const int row = (int)(i >> 9);           // E_ = 512
        const int e   = (int)(i & (E_ - 1));
        dst[i] = f2bf(emb[(size_t)tok[row] * E_ + e]);
    }
}

// ---------- WMMA GEMM: C[M,N] = A[M,K](bf16) * Bt[N,K]^T(bf16) + bias[N] ----------
// block = 128 threads (4 waves); each wave owns a 16x64 strip.
// grid = (N/256, M/16). M, N, K all multiples of tile sizes here.
// K loop is software-pipelined one deep so next fragments load under the WMMAs.
template <bool NT>
__global__ void k_gemm_bf16(const unsigned short* __restrict__ A,
                            const unsigned short* __restrict__ Bt,
                            const float* __restrict__ bias,
                            float* __restrict__ C, int K, int N) {
    const int wave = threadIdx.x >> 5;
    const int lane = threadIdx.x & 31;
    const int row0 = blockIdx.y << 4;
    const int col0 = (blockIdx.x << 8) + (wave << 6);

    v8f acc[4] = {v8f{}, v8f{}, v8f{}, v8f{}};

    v16bf a = frag_bf(A, K, row0, 0);
    v16bf b[4];
#pragma unroll
    for (int j = 0; j < 4; ++j) b[j] = frag_bf(Bt, K, col0 + (j << 4), 0);

    for (int k = 32; k < K; k += 32) {
        const v16bf an = frag_bf(A, K, row0, k);
        v16bf bn[4];
#pragma unroll
        for (int j = 0; j < 4; ++j) bn[j] = frag_bf(Bt, K, col0 + (j << 4), k);
#pragma unroll
        for (int j = 0; j < 4; ++j) acc[j] = wmma_bf16(a, b[j], acc[j]);
        a = an;
#pragma unroll
        for (int j = 0; j < 4; ++j) b[j] = bn[j];
    }
#pragma unroll
    for (int j = 0; j < 4; ++j) acc[j] = wmma_bf16(a, b[j], acc[j]);

    // D layout: VGPR i holds M = i (lanes 0-15) / M = i+8 (lanes 16-31), N = lane&15
    const int rb = row0 + ((lane >> 4) << 3);
#pragma unroll
    for (int j = 0; j < 4; ++j) {
        const int col = col0 + (j << 4) + (lane & 15);
        const float bv = bias[col];
#pragma unroll
        for (int i = 0; i < 8; ++i) {
            const float v = acc[j][i] + bv;
            float* dst = &C[(size_t)(rb + i) * N + col];
            if (NT) __builtin_nontemporal_store(v, dst);   // logits: written once, never read
            else    *dst = v;
        }
    }
}

// ---------- GRU step ----------
// grid = H/16 blocks, 96 threads (3 waves: z, r, h-candidate).
// Each wave: rec_tile[16,16] = h_prev[16,1024] @ Ut_rows + b_rec  via 32 WMMAs,
// then the block combines the three gate tiles from LDS.
// Recurrent state kept in parallel f32 (elementwise path) + bf16 (WMMA A) buffers.
__global__ void k_gru_step(const float* __restrict__ hp_f32,          // [16, H]
                           const unsigned short* __restrict__ hp_bf,  // [16, H] bf16
                           const unsigned short* __restrict__ Ut,     // [3H, H] bf16
                           const float* __restrict__ b_rec,           // [3H]
                           const float* __restrict__ xp,              // [B*T, 3H]
                           float* __restrict__ hn_f32,                // [16, H]
                           unsigned short* __restrict__ hn_bf,        // [16, H] bf16
                           unsigned short* __restrict__ hs_bf,        // [B*T, H] bf16
                           int t) {
    __shared__ float recs[3][16][16];
    const int wave = threadIdx.x >> 5;
    const int lane = threadIdx.x & 31;
    const int col0 = blockIdx.x << 4;
    const int brow = wave * H_ + col0;

    v8f acc = {};
    v16bf a = frag_bf(hp_bf, H_, 0, 0);
    v16bf b = frag_bf(Ut, H_, brow, 0);
    for (int k = 32; k < H_; k += 32) {
        const v16bf an = frag_bf(hp_bf, H_, 0, k);
        const v16bf bn = frag_bf(Ut, H_, brow, k);
        acc = wmma_bf16(a, b, acc);
        a = an;
        b = bn;
    }
    acc = wmma_bf16(a, b, acc);

    {
        const int n  = lane & 15;
        const int mb = (lane >> 4) << 3;
        const float bv = b_rec[brow + n];
#pragma unroll
        for (int i = 0; i < 8; ++i) recs[wave][mb + i][n] = acc[i] + bv;
    }
    __syncthreads();

    for (int idx = threadIdx.x; idx < 256; idx += blockDim.x) {
        const int m = idx >> 4, n = idx & 15;
        const int col = col0 + n;
        const size_t rrow = (size_t)(m * T_ + t) * H3;
        const float xz = xp[rrow + col];
        const float xr = xp[rrow + col + H_];
        const float xh = xp[rrow + col + 2 * H_];
        const float z  = 1.f / (1.f + __expf(-(xz + recs[0][m][n])));
        const float r  = 1.f / (1.f + __expf(-(xr + recs[1][m][n])));
        const float hh = tanhf(xh + r * recs[2][m][n]);
        const float hp = hp_f32[(size_t)m * H_ + col];
        const float hn = z * hp + (1.f - z) * hh;
        const unsigned short hb = f2bf(hn);
        hn_f32[(size_t)m * H_ + col] = hn;
        hn_bf[(size_t)m * H_ + col] = hb;
        hs_bf[(size_t)(m * T_ + t) * H_ + col] = hb;
    }
}

// ---------- host ----------

extern "C" void kernel_launch(void* const* d_in, const int* in_sizes, int n_in,
                              void* d_out, int out_size, void* d_ws, size_t ws_size,
                              hipStream_t stream) {
    (void)in_sizes; (void)n_in; (void)out_size; (void)ws_size;
    const int*   tok = (const int*)d_in[0];
    const float* h0  = (const float*)d_in[1];
    const float* emb = (const float*)d_in[2];
    const float* W   = (const float*)d_in[3];
    const float* U   = (const float*)d_in[4];
    const float* bb  = (const float*)d_in[5];     // [2, 3H]
    const float* Wo  = (const float*)d_in[6];
    const float* bo  = (const float*)d_in[7];
    float* out = (float*)d_out;

    char* ws = (char*)d_ws;
    size_t off = 0;
    auto carve = [&](size_t bytes) -> void* {
        void* p = ws + off;
        off = (off + bytes + 255) & ~(size_t)255;
        return p;
    };
    unsigned short* Wt   = (unsigned short*)carve((size_t)E_ * H3 * 2);   // [3H, E]
    unsigned short* Ut   = (unsigned short*)carve((size_t)H_ * H3 * 2);   // [3H, H]
    unsigned short* Wot  = (unsigned short*)carve((size_t)H_ * V_ * 2);   // [V, H]
    unsigned short* xbf  = (unsigned short*)carve((size_t)B_ * T_ * E_ * 2);
    float*          xp   = (float*)carve((size_t)B_ * T_ * H3 * 4);
    unsigned short* hsbf = (unsigned short*)carve((size_t)B_ * T_ * H_ * 2);
    float*          hb0  = (float*)carve((size_t)B_ * H_ * 4);
    float*          hb1  = (float*)carve((size_t)B_ * H_ * 4);
    unsigned short* hbfI = (unsigned short*)carve((size_t)B_ * H_ * 2);
    unsigned short* hbf0 = (unsigned short*)carve((size_t)B_ * H_ * 2);
    unsigned short* hbf1 = (unsigned short*)carve((size_t)B_ * H_ * 2);

    auto blocks = [](size_t total) { return (unsigned)((total + 255) / 256); };

    // weight conversion / transposition to bf16
    k_cvt_transpose<<<blocks((size_t)E_ * H3), 256, 0, stream>>>(W,  Wt,  E_, H3);
    k_cvt_transpose<<<blocks((size_t)H_ * H3), 256, 0, stream>>>(U,  Ut,  H_, H3);
    k_cvt_transpose<<<blocks((size_t)H_ * V_), 256, 0, stream>>>(Wo, Wot, H_, V_);
    k_embed<<<blocks((size_t)B_ * T_ * E_), 256, 0, stream>>>(tok, emb, xbf);
    k_cvt_bf16<<<blocks((size_t)B_ * H_), 256, 0, stream>>>(h0, hbfI, B_ * H_);

    // x_proj = x @ W + b[0]   -> [4000, 3072] f32  (re-read 250x: regular temporal)
    k_gemm_bf16<false><<<dim3(H3 / 256, (B_ * T_) / 16), 128, 0, stream>>>(
        xbf, Wt, bb, xp, E_, H3);

    // GRU recurrence over T (sequential launches; double-buffered f32+bf16 state)
    for (int t = 0; t < T_; ++t) {
        const float*          hpf = (t == 0) ? h0   : (((t - 1) & 1) ? hb1  : hb0);
        const unsigned short* hpb = (t == 0) ? hbfI : (((t - 1) & 1) ? hbf1 : hbf0);
        float*          hnf = (t & 1) ? hb1  : hb0;
        unsigned short* hnb = (t & 1) ? hbf1 : hbf0;
        k_gru_step<<<dim3(H_ / 16), 96, 0, stream>>>(
            hpf, hpb, Ut, bb + H3, xp, hnf, hnb, hsbf, t);
    }

    // logits = hs @ Wo + bo  -> [4000, 32000] f32  (streamed out non-temporally)
    k_gemm_bf16<true><<<dim3(V_ / 256, (B_ * T_) / 16), 128, 0, stream>>>(
        hsbf, Wot, bo, out, H_, V_);
}
